// MultiHeadAttention_63187558859261
// MI455X (gfx1250) — matrix operand order repaired and checked by reading, not compiled, not run on hardware
//
#include <hip/hip_runtime.h>
#include <hip/hip_bf16.h>

// ---------------------------------------------------------------------------
// CDNA5 (gfx1250) WMMA-based fused MHA-with-memory-compression.
// - All GEMMs via v_wmma_f32_16x16x32_bf16 (wave32), 128x64 block tiles,
//   8 waves x (2x2) 16x16 subtiles, 4 WMMA per wave per K-step.
// - LDS fragment reads are contiguous 16B-aligned runs (ds_load_b128).
// - bf16->bf16 tile staging uses GLOBAL_LOAD_ASYNC_TO_LDS_B128 when the
//   toolchain exposes it (ASYNCcnt + s_wait_asynccnt), else VGPR b128 copies.
// - fp32->bf16 staging is float4-vectorized with packed ds_store_b64.
// ---------------------------------------------------------------------------

typedef __attribute__((ext_vector_type(16))) __bf16          v16bf;
typedef __attribute__((ext_vector_type(16))) unsigned short  v16us;
typedef __attribute__((ext_vector_type(8)))  unsigned short  v8us;
typedef __attribute__((ext_vector_type(8)))  float           v8f;
typedef int v4i_native __attribute__((vector_size(16)));

#if defined(__HIP_DEVICE_COMPILE__) && \
    __has_builtin(__builtin_amdgcn_global_load_async_to_lds_b128)
#define HAVE_ASYNC_LDS 1
#else
#define HAVE_ASYNC_LDS 0
#endif

__device__ __forceinline__ unsigned short f2bf(float f) {
  unsigned u = __builtin_bit_cast(unsigned, f);
  u += 0x7FFFu + ((u >> 16) & 1u);        // round-to-nearest-even
  return (unsigned short)(u >> 16);
}
__device__ __forceinline__ unsigned pk2bf(float a, float b) {
  return (unsigned)f2bf(a) | ((unsigned)f2bf(b) << 16);
}

#if HAVE_ASYNC_LDS
__device__ __forceinline__ void async_cp_b128(const unsigned short* g,
                                              unsigned short* l) {
  __builtin_amdgcn_global_load_async_to_lds_b128(
      (__attribute__((address_space(1))) v4i_native*)(unsigned short*)g,
      (__attribute__((address_space(3))) v4i_native*)l, 0, 0);
}
__device__ __forceinline__ void async_fence() {
#if __has_builtin(__builtin_amdgcn_s_wait_asynccnt)
  __builtin_amdgcn_s_wait_asynccnt(0);
#else
  asm volatile("s_wait_asynccnt 0" ::: "memory");
#endif
}
#else
__device__ __forceinline__ void async_cp_b128(const unsigned short* g,
                                              unsigned short* l) {
  *(v8us*)l = *(const v8us*)g;            // global_load_b128 + ds_store_b128
}
__device__ __forceinline__ void async_fence() {}
#endif

#define BM 128
#define BN 64
#define BK 32
// LDS row stride in shorts: 40 shorts = 80 B (16B-aligned rows; 20-bank step
// => 16 consecutive rows hit 16 distinct banks).
#define ASTR 40
#define BTSTR 40

// Build a v16bf fragment from two aligned 8-short LDS runs.
__device__ __forceinline__ v16bf frag_from_lds(const unsigned short* p0,
                                               const unsigned short* p1) {
  v8us lo = *(const v8us*)p0;
  v8us hi = *(const v8us*)p1;
  v16us u = __builtin_shufflevector(lo, hi, 0, 1, 2, 3, 4, 5, 6, 7,
                                            8, 9, 10, 11, 12, 13, 14, 15);
  return __builtin_bit_cast(v16bf, u);
}

// C[M,N] = alpha * A[M,K] * op(B) + bias[n] + addv[n]
template<bool A_F32, bool B_F32, bool C_F32, bool TRANSB>
__global__ __launch_bounds__(256) void gemm_wmma_kernel(
    const void* __restrict__ Ap, const void* __restrict__ Bp, void* __restrict__ Cp,
    int M, int N, int K, int lda, int ldb, int ldc,
    float alpha, const float* __restrict__ bias, const float* __restrict__ addv)
{
  (void)M; (void)N;   // all dims are exact tile multiples in this problem
  __shared__ __align__(16) unsigned short As[BM * ASTR];   // A tile (m,k)
  __shared__ __align__(16) unsigned short Bt[BN * BTSTR];  // B tile TRANSPOSED (n,k)

  const int tid  = threadIdx.x;
  const int lane = tid & 31;
  const int wave = tid >> 5;
  const int hf   = lane >> 4;    // wave32 half
  const int l16  = lane & 15;

  const int gm0 = blockIdx.y * BM;
  const int gn0 = blockIdx.x * BN;

  const int wm = (wave & 3) * 32;   // wave's 32-row strip
  const int wn = (wave >> 2) * 32;  // wave's 32-col strip

  v8f acc00 = {}, acc01 = {}, acc10 = {}, acc11 = {};

  const float*          Af = (const float*)Ap;
  const unsigned short* Ab = (const unsigned short*)Ap;
  const float*          Bf = (const float*)Bp;
  const unsigned short* Bb = (const unsigned short*)Bp;

  for (int k0 = 0; k0 < K; k0 += BK) {
    // ================= stage A (128x32) =================
    if (A_F32) {
      // float4 loads, pack 4 bf16, ds_store_b64. 128 rows x 8 quads = 1024.
      #pragma unroll
      for (int e = 0; e < 4; ++e) {
        int idx = e * 256 + tid;
        int r = idx >> 3, c4 = (idx & 7) * 4;
        const float4 f = *(const float4*)(Af + (size_t)(gm0 + r) * lda + (k0 + c4));
        *(uint2*)&As[r * ASTR + c4] = make_uint2(pk2bf(f.x, f.y), pk2bf(f.z, f.w));
      }
    } else {
      // straight bf16 copy: 128 rows x 4 16B-chunks = 512 -> 2 per thread.
      #pragma unroll
      for (int e = 0; e < 2; ++e) {
        int idx = e * 256 + tid;
        int r = idx >> 2, c8 = (idx & 3) * 8;
        async_cp_b128(Ab + (size_t)(gm0 + r) * lda + (k0 + c8), &As[r * ASTR + c8]);
      }
    }
    // ================= stage B (32x64) -> Bt[n][k] =================
    if (TRANSB) {
      if (B_F32) {
        // row n contiguous in k: float4 -> packed b64. 64 x 8 = 512.
        #pragma unroll
        for (int e = 0; e < 2; ++e) {
          int idx = e * 256 + tid;
          int n = idx >> 3, c4 = (idx & 7) * 4;
          const float4 f = *(const float4*)(Bf + (size_t)(gn0 + n) * ldb + (k0 + c4));
          *(uint2*)&Bt[n * BTSTR + c4] = make_uint2(pk2bf(f.x, f.y), pk2bf(f.z, f.w));
        }
      } else {
        // bf16 copy, contiguous in k: 64 rows x 4 16B-chunks = 256 -> 1 each.
        int n = tid >> 2, c8 = (tid & 3) * 8;
        async_cp_b128(Bb + (size_t)(gn0 + n) * ldb + (k0 + c8), &Bt[n * BTSTR + c8]);
      }
    } else {
      // global rows are k-major: gather 4 k's per n, pack, ds_store_b64.
      // 8 k-quads x 64 n = 512 -> 2 per thread; loads coalesced over n.
      #pragma unroll
      for (int e = 0; e < 2; ++e) {
        int idx = e * 256 + tid;
        int k4 = (idx >> 6) * 4, n = idx & 63;
        unsigned p0, p1;
        if (B_F32) {
          float f0 = Bf[(size_t)(k0 + k4 + 0) * ldb + (gn0 + n)];
          float f1 = Bf[(size_t)(k0 + k4 + 1) * ldb + (gn0 + n)];
          float f2 = Bf[(size_t)(k0 + k4 + 2) * ldb + (gn0 + n)];
          float f3 = Bf[(size_t)(k0 + k4 + 3) * ldb + (gn0 + n)];
          p0 = pk2bf(f0, f1); p1 = pk2bf(f2, f3);
        } else {
          unsigned short s0 = Bb[(size_t)(k0 + k4 + 0) * ldb + (gn0 + n)];
          unsigned short s1 = Bb[(size_t)(k0 + k4 + 1) * ldb + (gn0 + n)];
          unsigned short s2 = Bb[(size_t)(k0 + k4 + 2) * ldb + (gn0 + n)];
          unsigned short s3 = Bb[(size_t)(k0 + k4 + 3) * ldb + (gn0 + n)];
          p0 = (unsigned)s0 | ((unsigned)s1 << 16);
          p1 = (unsigned)s2 | ((unsigned)s3 << 16);
        }
        *(uint2*)&Bt[n * BTSTR + k4] = make_uint2(p0, p1);
      }
    }
    if (!A_F32 || (TRANSB && !B_F32)) async_fence();
    __syncthreads();

    // ---- A fragments: K runs [hf*8 .. +7] and [16+hf*8 .. +7] per lane ----
    const unsigned short* ar0 = &As[(wm + l16) * ASTR + hf * 8];
    const unsigned short* ar1 = &As[(wm + 16 + l16) * ASTR + hf * 8];
    v16bf a0 = frag_from_lds(ar0, ar0 + 16);
    v16bf a1 = frag_from_lds(ar1, ar1 + 16);
    // ---- B fragments: lane = n, K runs [hf*8 .. +7] and [16+hf*8 .. +7] ----
    const unsigned short* br0 = &Bt[(wn + l16) * BTSTR + hf * 8];
    const unsigned short* br1 = &Bt[(wn + 16 + l16) * BTSTR + hf * 8];
    v16bf b0 = frag_from_lds(br0, br0 + 16);
    v16bf b1 = frag_from_lds(br1, br1 + 16);

    acc00 = __builtin_amdgcn_wmma_f32_16x16x32_bf16(false, a0, false, b0,
                                                    (short)0, acc00, false, false);
    acc01 = __builtin_amdgcn_wmma_f32_16x16x32_bf16(false, a0, false, b1,
                                                    (short)0, acc01, false, false);
    acc10 = __builtin_amdgcn_wmma_f32_16x16x32_bf16(false, a1, false, b0,
                                                    (short)0, acc10, false, false);
    acc11 = __builtin_amdgcn_wmma_f32_16x16x32_bf16(false, a1, false, b1,
                                                    (short)0, acc11, false, false);
    __syncthreads();
  }

  // ---- epilogue: C/D layout = 8 VGPRs, lane = n, hf selects M base 0/8 ----
  const int colA = gn0 + wn + l16;
  const int colB = colA + 16;
  float addA = 0.0f, addB = 0.0f;
  if (bias) { addA += bias[colA]; addB += bias[colB]; }
  if (addv) { addA += addv[colA]; addB += addv[colB]; }

  #pragma unroll
  for (int sub = 0; sub < 2; ++sub) {       // m-subtile (wm / wm+16)
    const v8f& ac0 = sub ? acc10 : acc00;
    const v8f& ac1 = sub ? acc11 : acc01;
    int row0 = gm0 + wm + sub * 16 + hf * 8;
    size_t baseA = (size_t)row0 * (size_t)ldc + (size_t)colA;
    size_t baseB = (size_t)row0 * (size_t)ldc + (size_t)colB;
    #pragma unroll
    for (int r = 0; r < 8; ++r) {
      float v0 = alpha * ac0[r] + addA;
      float v1 = alpha * ac1[r] + addB;
      if (C_F32) {
        ((float*)Cp)[baseA] = v0;
        ((float*)Cp)[baseB] = v1;
      } else {
        ((unsigned short*)Cp)[baseA] = f2bf(v0);
        ((unsigned short*)Cp)[baseB] = f2bf(v1);
      }
      baseA += (size_t)ldc;
      baseB += (size_t)ldc;
    }
  }
}

// In-place softmax over rows of length 1024 (one block per row).
__global__ __launch_bounds__(256) void softmax_rows_kernel(float* __restrict__ attn)
{
  __shared__ float red[256];
  float* p = attn + (size_t)blockIdx.x * 1024;
  int tid = threadIdx.x;
  float v0 = p[tid], v1 = p[tid + 256], v2 = p[tid + 512], v3 = p[tid + 768];
  float mx = fmaxf(fmaxf(v0, v1), fmaxf(v2, v3));
  red[tid] = mx; __syncthreads();
  for (int s = 128; s > 0; s >>= 1) {
    if (tid < s) red[tid] = fmaxf(red[tid], red[tid + s]);
    __syncthreads();
  }
  float m = red[0]; __syncthreads();
  v0 = __expf(v0 - m); v1 = __expf(v1 - m); v2 = __expf(v2 - m); v3 = __expf(v3 - m);
  red[tid] = v0 + v1 + v2 + v3; __syncthreads();
  for (int s = 128; s > 0; s >>= 1) {
    if (tid < s) red[tid] += red[tid + s];
    __syncthreads();
  }
  float inv = 1.0f / red[0];
  p[tid] = v0 * inv; p[tid + 256] = v1 * inv; p[tid + 512] = v2 * inv; p[tid + 768] = v3 * inv;
}

// mask_c[b, n] = mask[b, 4n] * -1e9  (pre-scaled additive logit bias)
__global__ void prep_mask_kernel(const float* __restrict__ mask, float* __restrict__ maskc)
{
  int i = blockIdx.x * blockDim.x + threadIdx.x;
  if (i < 2048) {
    int b = i >> 10, n = i & 1023;
    maskc[i] = mask[b * 4096 + n * 4] * -1.0e9f;
  }
}

extern "C" void kernel_launch(void* const* d_in, const int* in_sizes, int n_in,
                              void* d_out, int out_size, void* d_ws, size_t ws_size,
                              hipStream_t stream) {
  (void)in_sizes; (void)n_in; (void)out_size; (void)ws_size;
  const float* v    = (const float*)d_in[0];
  const float* k    = (const float*)d_in[1];
  const float* q    = (const float*)d_in[2];
  const float* mask = (const float*)d_in[3];
  const float* wq_w = (const float*)d_in[4];
  const float* wq_b = (const float*)d_in[5];
  const float* wk_w = (const float*)d_in[6];
  const float* wk_b = (const float*)d_in[7];
  const float* wv_w = (const float*)d_in[8];
  const float* wv_b = (const float*)d_in[9];
  const float* kc_w = (const float*)d_in[10];
  const float* kc_b = (const float*)d_in[11];
  const float* vc_w = (const float*)d_in[12];
  const float* vc_b = (const float*)d_in[13];
  const float* wo_w = (const float*)d_in[14];
  const float* wo_b = (const float*)d_in[15];

  const int Bn = 2, L = 4096, D = 1024, H = 16, Lc = 1024; // Lc = L / MEM_COMP

  float* out  = (float*)d_out;                      // (B, L, D)
  float* attn = out + (size_t)Bn * L * D;           // (B, H, L, Lc) fp32 output

  // bf16 intermediates in workspace (~72 MB)
  unsigned short* qh  = (unsigned short*)d_ws;                  // (B, L, D)
  unsigned short* kh  = qh  + (size_t)Bn * L  * D;              // (B, L, D)
  unsigned short* vh  = kh  + (size_t)Bn * L  * D;              // (B, L, D)
  unsigned short* khc = vh  + (size_t)Bn * L  * D;              // (B, Lc, D)
  unsigned short* vhc = khc + (size_t)Bn * Lc * D;              // (B, Lc, D)
  unsigned short* ctx = vhc + (size_t)Bn * Lc * D;              // (B, L, D)
  float* maskc = (float*)(ctx + (size_t)Bn * L * D);            // (B, Lc)

  dim3 blk(256);

  // 1) Input projections: qh = q@Wq + bq, etc.  (8192x1024x1024)
  {
    dim3 grid(D / BN, (Bn * L) / BM);
    gemm_wmma_kernel<true, true, false, false><<<grid, blk, 0, stream>>>(
        q, wq_w, qh, Bn * L, D, D, D, D, D, 1.0f, wq_b, nullptr);
    gemm_wmma_kernel<true, true, false, false><<<grid, blk, 0, stream>>>(
        k, wk_w, kh, Bn * L, D, D, D, D, D, 1.0f, wk_b, nullptr);
    gemm_wmma_kernel<true, true, false, false><<<grid, blk, 0, stream>>>(
        v, wv_w, vh, Bn * L, D, D, D, D, D, 1.0f, wv_b, nullptr);
  }

  // 2) Strided conv (window=4, stride=4, SAME -> exact) as one GEMM:
  //    khc[b,t,o] = sum_{w,i} kh[b,4t+w,i]*kc_w[w,i,o] == (2048x4096)@(4096x1024)
  {
    dim3 grid(D / BN, (Bn * Lc) / BM);
    gemm_wmma_kernel<false, true, false, false><<<grid, blk, 0, stream>>>(
        kh, kc_w, khc, Bn * Lc, D, 4 * D, 4 * D, D, D, 1.0f, kc_b, nullptr);
    gemm_wmma_kernel<false, true, false, false><<<grid, blk, 0, stream>>>(
        vh, vc_w, vhc, Bn * Lc, D, 4 * D, 4 * D, D, D, 1.0f, vc_b, nullptr);
  }

  // 3) mask_c = mask[..., ::4] * -1e9
  prep_mask_kernel<<<dim3(8), blk, 0, stream>>>(mask, maskc);

  // 4) Logits per (b,h): attn = 1/8 * Qh @ Khc^T + mask_c   (4096 x 1024 x 64)
  for (int z = 0; z < Bn * H; ++z) {
    int b = z / H, h = z % H;
    dim3 grid(Lc / BN, L / BM);
    gemm_wmma_kernel<false, false, true, true><<<grid, blk, 0, stream>>>(
        qh  + (size_t)b * L  * D + h * 64,
        khc + (size_t)b * Lc * D + h * 64,
        attn + (size_t)z * L * Lc,
        L, Lc, 64, D, D, Lc, 0.125f, nullptr, maskc + (size_t)b * Lc);
  }

  // 5) Softmax rows (B*H*L rows of 1024), in place in the attn output region.
  softmax_rows_kernel<<<dim3(Bn * H * L), blk, 0, stream>>>(attn);

  // 6) Context per (b,h): ctx = attn @ Vhc  (4096 x 64 x 1024), stored (B,L,D)
  for (int z = 0; z < Bn * H; ++z) {
    int b = z / H, h = z % H;
    dim3 grid(64 / BN, L / BM);
    gemm_wmma_kernel<true, false, false, false><<<grid, blk, 0, stream>>>(
        attn + (size_t)z * L * Lc,
        vhc  + (size_t)b * Lc * D + h * 64,
        ctx  + (size_t)b * L  * D + h * 64,
        L, 64, Lc, Lc, D, D, 1.0f, nullptr, nullptr);
  }

  // 7) Output projection: out = ctx @ Wo + bo  (8192 x 1024 x 1024)
  {
    dim3 grid(D / BN, (Bn * L) / BM);
    gemm_wmma_kernel<false, true, true, false><<<grid, blk, 0, stream>>>(
        ctx, wo_w, out, Bn * L, D, D, D, D, D, 1.0f, wo_b, nullptr);
  }
}
